// AttentionBlock_37056977829993
// MI455X (gfx1250) — compile-verified, matrix operand order checked by reading, not run on hardware
//
#include <hip/hip_runtime.h>
#include <cstdint>

typedef __attribute__((ext_vector_type(16))) _Float16 v16h;
typedef __attribute__((ext_vector_type(8)))  float    v8f;

#define BB   16
#define CC   512
#define NN   1024
#define C3   1536
#define CPG  16      // channels per group (512 / 32 groups)

// K-pair base within a 16x32 f16 A/B fragment (wave32 layout, ISA 7.12.2):
// VGPR j (0..7): k = (j<4?0:16) + half*8 + (j&3)*2
// j=0..3 and j=4..7 each form 16 contiguous bytes -> compiler merges to global_load_b128.
__device__ __forceinline__ int kbase(int j, int half) {
  return ((j & 4) << 2) + half * 8 + ((j & 3) << 1);
}

union FragH { v16h v; uint32_t u[8]; _Float16 h[16]; };

// All operands contiguous-in-K:
//   A(m,k) = A[m*lda + k]   (row-major)
//   B(k,n) = B[n*ldb + k]   (n-major)
// Each wave computes a 32x32 output region (2x2 WMMA tiles) -> 2x fragment reuse.
__device__ __forceinline__ void wave_gemm_2x2(const _Float16* __restrict__ A, int lda,
                                              const _Float16* __restrict__ B, int ldb,
                                              int K, int m0, int n0, v8f acc[2][2]) {
  const int lane = threadIdx.x & 31;
  const int half = lane >> 4;
  const int r    = lane & 15;
  const _Float16* Ap0 = A + (size_t)(m0 + r)      * lda;
  const _Float16* Ap1 = A + (size_t)(m0 + 16 + r) * lda;
  const _Float16* Bp0 = B + (size_t)(n0 + r)      * ldb;
  const _Float16* Bp1 = B + (size_t)(n0 + 16 + r) * ldb;
  for (int k0 = 0; k0 < K; k0 += 32) {
    FragH a0, a1, b0, b1;
#pragma unroll
    for (int j = 0; j < 8; ++j) {
      const int kb = k0 + kbase(j, half);
      a0.u[j] = *(const uint32_t*)(Ap0 + kb);
      a1.u[j] = *(const uint32_t*)(Ap1 + kb);
      b0.u[j] = *(const uint32_t*)(Bp0 + kb);
      b1.u[j] = *(const uint32_t*)(Bp1 + kb);
    }
    if (k0 + 32 < K) {   // cheap latency hint for the next K slice
      __builtin_prefetch(Ap0 + k0 + 32, 0, 1);
      __builtin_prefetch(Bp0 + k0 + 32, 0, 1);
    }
    acc[0][0] = __builtin_amdgcn_wmma_f32_16x16x32_f16(false, a0.v, false, b0.v, (short)0, acc[0][0], false, false);
    acc[0][1] = __builtin_amdgcn_wmma_f32_16x16x32_f16(false, a0.v, false, b1.v, (short)0, acc[0][1], false, false);
    acc[1][0] = __builtin_amdgcn_wmma_f32_16x16x32_f16(false, a1.v, false, b0.v, (short)0, acc[1][0], false, false);
    acc[1][1] = __builtin_amdgcn_wmma_f32_16x16x32_f16(false, a1.v, false, b1.v, (short)0, acc[1][1], false, false);
  }
}

// ---------------- prep kernels ----------------

__global__ void __launch_bounds__(256) cvt_f32_to_f16(const float* __restrict__ in,
                                                      _Float16* __restrict__ out, int n) {
  int i = blockIdx.x * blockDim.x + threadIdx.x;
  if (i < n) out[i] = (_Float16)in[i];
}

// one block per (batch, group): reduce 16 channels * 1024 pixels (contiguous 16384 floats)
__global__ void __launch_bounds__(256) gn_stats_kernel(const float* __restrict__ x,
                                                       float* __restrict__ stats) {
  __shared__ float s1[256], s2[256];
  const int bg = blockIdx.x;
  const float* xp = x + (size_t)bg * (CPG * NN);
  float a = 0.f, b = 0.f;
  for (int i = threadIdx.x; i < CPG * NN; i += 256) { float v = xp[i]; a += v; b += v * v; }
  s1[threadIdx.x] = a; s2[threadIdx.x] = b;
  __syncthreads();
  for (int s = 128; s > 0; s >>= 1) {
    if (threadIdx.x < s) { s1[threadIdx.x] += s1[threadIdx.x + s];
                           s2[threadIdx.x] += s2[threadIdx.x + s]; }
    __syncthreads();
  }
  if (threadIdx.x == 0) {
    const float inv = 1.0f / (float)(CPG * NN);
    float mean = s1[0] * inv;
    float var  = s2[0] * inv - mean * mean;
    stats[2 * bg]     = mean;
    stats[2 * bg + 1] = rsqrtf(var + 1e-5f);
  }
}

// groupnorm + f16, written TRANSPOSED: Ht[b][n][c]  (pixel-major, so GEMM-B is contiguous in K)
__global__ void __launch_bounds__(256) gn_apply_t_kernel(const float* __restrict__ x,
    const float* __restrict__ gamma, const float* __restrict__ beta,
    const float* __restrict__ stats, _Float16* __restrict__ Ht) {
  const size_t i4 = ((size_t)blockIdx.x * blockDim.x + threadIdx.x) * 4;  // index in [b][c][n]
  const int n  = (int)(i4 & (NN - 1));
  const int c  = (int)((i4 >> 10) & (CC - 1));
  const int bg = (int)(i4 >> 14);              // = b*32 + c/16
  const int b  = bg >> 5;
  const float4 xv = *(const float4*)(x + i4);
  const float mean = stats[2 * bg], rstd = stats[2 * bg + 1];
  const float g = gamma[c] * rstd, bt = beta[c] - mean * gamma[c] * rstd;
  _Float16* hp = Ht + ((size_t)b * NN + n) * CC + c;   // scatter along n (stride CC)
  hp[0 * CC] = (_Float16)(xv.x * g + bt);
  hp[1 * CC] = (_Float16)(xv.y * g + bt);
  hp[2 * CC] = (_Float16)(xv.z * g + bt);
  hp[3 * CC] = (_Float16)(xv.w * g + bt);
}

// ---------------- GEMM kernels (32x32 per wave, 4 waves/block) ----------------

// QKV[o][n] = sum_c W[o][c] * Ht[n][c] + bias[o]   (M=1536, N=1024, K=512)
// epilogue: Q,K stored pixel-major (Qt/Kt[n][c]), V stored channel-major (V[c][n])
__global__ void __launch_bounds__(128) qkv_gemm_kernel(const _Float16* __restrict__ W,
    const _Float16* __restrict__ Ht, const float* __restrict__ bias,
    _Float16* __restrict__ Qt, _Float16* __restrict__ Kt, _Float16* __restrict__ Vv) {
  const int t  = blockIdx.x * 4 + (threadIdx.x >> 5);
  const int m0 = (t % (C3 / 32)) * 32, n0 = (t / (C3 / 32)) * 32;
  v8f acc[2][2] = {};
  wave_gemm_2x2(W, CC, Ht, CC, CC, m0, n0, acc);
  const int lane = threadIdx.x & 31, half = lane >> 4, r = lane & 15;
#pragma unroll
  for (int mi = 0; mi < 2; ++mi) {
#pragma unroll
    for (int ni = 0; ni < 2; ++ni) {
      const int mb = m0 + mi * 16;          // uniform per 16-tile
      const int n  = n0 + ni * 16 + r;
      const int mbase = mb + 8 * half;      // bias row for element v: mbase + v
      const int seg = mb >> 9;              // 0=Q, 1=K, 2=V
      if (seg < 2) {                        // transposed store: [n][c], contiguous along m
        _Float16* dst = (seg == 0 ? Qt : Kt) + (size_t)n * CC + ((mbase) & (CC - 1));
#pragma unroll
        for (int v = 0; v < 8; v += 2) {
          union { _Float16 h[2]; uint32_t u; } pk;
          pk.h[0] = (_Float16)(acc[mi][ni][v]     + bias[mbase + v]);
          pk.h[1] = (_Float16)(acc[mi][ni][v + 1] + bias[mbase + v + 1]);
          *(uint32_t*)(dst + v) = pk.u;
        }
      } else {                              // V kept channel-major: [c][n]
#pragma unroll
        for (int v = 0; v < 8; ++v)
          Vv[(size_t)(mbase - 1024 + v) * NN + n] =
              (_Float16)(acc[mi][ni][v] + bias[mbase + v]);
      }
    }
  }
}

// S[i][j] = scale * sum_c Qt[i][c] * Kt[j][c]   (M=N=1024, K=512), f32 out
__global__ void __launch_bounds__(128) scores_kernel(const _Float16* __restrict__ Qt,
    const _Float16* __restrict__ Kt, float* __restrict__ S) {
  const int t  = blockIdx.x * 4 + (threadIdx.x >> 5);
  const int m0 = (t & 31) * 32, n0 = (t >> 5) * 32;
  v8f acc[2][2] = {};
  wave_gemm_2x2(Qt, CC, Kt, CC, CC, m0, n0, acc);
  const float scale = 0.04419417382415922f;   // 1/sqrt(512)
  const int lane = threadIdx.x & 31, half = lane >> 4, r = lane & 15;
#pragma unroll
  for (int mi = 0; mi < 2; ++mi)
#pragma unroll
    for (int ni = 0; ni < 2; ++ni) {
      const int mb = m0 + mi * 16 + 8 * half, n = n0 + ni * 16 + r;
#pragma unroll
      for (int v = 0; v < 8; ++v)
        S[(size_t)(mb + v) * NN + n] = acc[mi][ni][v] * scale;
    }
}

// row softmax over 1024 entries, f32 in -> f16 probs out (P[i][j], i-major)
__global__ void __launch_bounds__(256) softmax_kernel(const float* __restrict__ S,
                                                      _Float16* __restrict__ P) {
  __shared__ float red[256];
  const int row = blockIdx.x;
  const int base = threadIdx.x * 4;
  float4 sv = *(const float4*)(S + (size_t)row * NN + base);
  float v0 = sv.x, v1 = sv.y, v2 = sv.z, v3 = sv.w;
  float m = fmaxf(fmaxf(v0, v1), fmaxf(v2, v3));
  red[threadIdx.x] = m; __syncthreads();
  for (int s = 128; s > 0; s >>= 1) {
    if (threadIdx.x < s) red[threadIdx.x] = fmaxf(red[threadIdx.x], red[threadIdx.x + s]);
    __syncthreads();
  }
  m = red[0]; __syncthreads();
  v0 = __expf(v0 - m); v1 = __expf(v1 - m); v2 = __expf(v2 - m); v3 = __expf(v3 - m);
  red[threadIdx.x] = v0 + v1 + v2 + v3; __syncthreads();
  for (int s = 128; s > 0; s >>= 1) {
    if (threadIdx.x < s) red[threadIdx.x] += red[threadIdx.x + s];
    __syncthreads();
  }
  const float inv = 1.0f / red[0];
  union { _Float16 h[4]; uint2 u; } o;
  o.h[0] = (_Float16)(v0 * inv); o.h[1] = (_Float16)(v1 * inv);
  o.h[2] = (_Float16)(v2 * inv); o.h[3] = (_Float16)(v3 * inv);
  *(uint2*)(P + (size_t)row * NN + base) = o.u;
}

// O[c][i] = sum_j V[c][j] * P[i][j]  (M=512, N=1024, K=1024); epilogue stores Ot[i][c] (pixel-major)
__global__ void __launch_bounds__(128) av_gemm_kernel(const _Float16* __restrict__ Vv,
    const _Float16* __restrict__ P, _Float16* __restrict__ Ot) {
  const int t  = blockIdx.x * 4 + (threadIdx.x >> 5);
  const int m0 = (t & 15) * 32, n0 = (t >> 4) * 32;
  v8f acc[2][2] = {};
  wave_gemm_2x2(Vv, NN, P, NN, NN, m0, n0, acc);
  const int lane = threadIdx.x & 31, half = lane >> 4, r = lane & 15;
#pragma unroll
  for (int mi = 0; mi < 2; ++mi)
#pragma unroll
    for (int ni = 0; ni < 2; ++ni) {
      const int mb = m0 + mi * 16 + 8 * half, n = n0 + ni * 16 + r;
      _Float16* dst = Ot + (size_t)n * CC + mb;    // contiguous along m
#pragma unroll
      for (int v = 0; v < 8; v += 2) {
        union { _Float16 h[2]; uint32_t u; } pk;
        pk.h[0] = (_Float16)acc[mi][ni][v];
        pk.h[1] = (_Float16)acc[mi][ni][v + 1];
        *(uint32_t*)(dst + v) = pk.u;
      }
    }
}

// out[o][n] = x[o][n] + bias[o] + sum_c Wp[o][c] * Ot[n][c]  (M=512, N=1024, K=512), f32 out
__global__ void __launch_bounds__(128) proj_kernel(const _Float16* __restrict__ Wp,
    const _Float16* __restrict__ Ot, const float* __restrict__ bias,
    const float* __restrict__ xb, float* __restrict__ out) {
  const int t  = blockIdx.x * 4 + (threadIdx.x >> 5);
  const int m0 = (t & 15) * 32, n0 = (t >> 4) * 32;
  v8f acc[2][2] = {};
  wave_gemm_2x2(Wp, CC, Ot, CC, CC, m0, n0, acc);
  const int lane = threadIdx.x & 31, half = lane >> 4, r = lane & 15;
#pragma unroll
  for (int mi = 0; mi < 2; ++mi)
#pragma unroll
    for (int ni = 0; ni < 2; ++ni) {
      const int mb = m0 + mi * 16 + 8 * half, n = n0 + ni * 16 + r;
#pragma unroll
      for (int v = 0; v < 8; ++v) {
        const size_t idx = (size_t)(mb + v) * NN + n;
        out[idx] = xb[idx] + bias[mb + v] + acc[mi][ni][v];
      }
    }
}

// ---------------- host launch ----------------

extern "C" void kernel_launch(void* const* d_in, const int* in_sizes, int n_in,
                              void* d_out, int out_size, void* d_ws, size_t ws_size,
                              hipStream_t stream) {
  const float* x      = (const float*)d_in[0];
  const float* gamma  = (const float*)d_in[1];
  const float* beta   = (const float*)d_in[2];
  const float* w_qkv  = (const float*)d_in[3];
  const float* b_qkv  = (const float*)d_in[4];
  const float* w_proj = (const float*)d_in[5];
  const float* b_proj = (const float*)d_in[6];
  float* out = (float*)d_out;

  char* ws = (char*)d_ws;
  // workspace layout (256B aligned), ~29.4 MB total (fully L2-resident)
  float*    stats  = (float*)   (ws + 0);           //   4 KB
  _Float16* Wqkvh  = (_Float16*)(ws + 4096);        // 1.5 MB   [1536][512]
  _Float16* Wprojh = (_Float16*)(ws + 1576960);     // 0.5 MB   [512][512]
  _Float16* Ht     = (_Float16*)(ws + 2101248);     // 16.8 MB  [B][N][C] pixel-major
  _Float16* Qt     = (_Float16*)(ws + 18878464);    // 1 MB     [N][C] per-batch
  _Float16* Kt     = (_Float16*)(ws + 19927040);    // 1 MB     [N][C] per-batch
  _Float16* Vv     = (_Float16*)(ws + 20975616);    // 1 MB     [C][N] per-batch
  float*    S      = (float*)   (ws + 22024192);    // 4 MB     [N][N] per-batch
  _Float16* P      = (_Float16*)(ws + 26218496);    // 2 MB     [N][N] per-batch
  _Float16* Ot     = (_Float16*)(ws + 28315648);    // 1 MB     [N][C] per-batch

  cvt_f32_to_f16<<<(C3 * CC) / 256, 256, 0, stream>>>(w_qkv, Wqkvh, C3 * CC);
  cvt_f32_to_f16<<<(CC * CC) / 256, 256, 0, stream>>>(w_proj, Wprojh, CC * CC);
  gn_stats_kernel<<<BB * 32, 256, 0, stream>>>(x, stats);
  gn_apply_t_kernel<<<(BB * CC * NN) / (256 * 4), 256, 0, stream>>>(x, gamma, beta, stats, Ht);

  for (int b = 0; b < BB; ++b) {
    const _Float16* Htb = Ht + (size_t)b * NN * CC;
    // 48x32 super-tiles of 32x32, 4 waves/block
    qkv_gemm_kernel<<<(48 * 32) / 4, 128, 0, stream>>>(Wqkvh, Htb, b_qkv, Qt, Kt, Vv);
    // 32x32 super-tiles
    scores_kernel<<<(32 * 32) / 4, 128, 0, stream>>>(Qt, Kt, S);
    softmax_kernel<<<NN, 256, 0, stream>>>(S, P);
    // 16x32 super-tiles
    av_gemm_kernel<<<(16 * 32) / 4, 128, 0, stream>>>(Vv, P, Ot);
    proj_kernel<<<(16 * 32) / 4, 128, 0, stream>>>(Wprojh, Ot, b_proj,
                                                   x + (size_t)b * CC * NN,
                                                   out + (size_t)b * CC * NN);
  }
}
